// SemanticVQ_68418829025874
// MI455X (gfx1250) — matrix operand order repaired and checked by reading, not compiled, not run on hardware
//
#include <hip/hip_runtime.h>
#include <hip/hip_bf16.h>

// ---------------- problem constants ----------------
#define KCODES 4096
#define DIM    512
#define NROWS  32768          // B*T = 8*4096
#define BM     64             // x rows per block in the WMMA kernel (4 tiles/wave)
#define DPAD   520            // padded LDS row stride (bf16 elems)

// dynamic LDS partition (bytes)
#define SM_X     (BM * DPAD * 2)                  // 66560
#define SM_CBSQ  (KCODES * 4)                     // 16384
#define SM_RBEST (8 * BM * 4)                     // 2048
#define SM_RIDX  (8 * BM * 4)                     // 2048
#define SM_TOTAL (SM_X + SM_CBSQ + SM_RBEST + SM_RIDX)   // 87040

typedef __attribute__((ext_vector_type(16))) __bf16 v16bf;
typedef __attribute__((ext_vector_type(8)))  __bf16 bf16x8;
typedef __attribute__((ext_vector_type(8)))  float  v8f;

// issue two plain 128-bit loads for one 16x32 B chunk (byte offsets are immediates)
#define TR16_ISSUE2(dA, dB, base, byteoff)                                   \
    asm volatile("global_load_b128 %0, %2, off offset:%c3\n\t"               \
                 "global_load_b128 %1, %2, off offset:%c4"                   \
                 : "=&v"(dA), "=&v"(dB)                                      \
                 : "v"(base), "i"(byteoff), "i"((byteoff) + 32)              \
                 : "memory")

// wait until <= n vmem loads outstanding; data-tie the buffers so WMMA
// consumers are ordered after the wait
#define TR16_WAIT2(dA, dB, nstr)                                             \
    asm volatile("s_wait_loadcnt " nstr : "+v"(dA), "+v"(dB) :: "memory")

#define CAT16(lo, hi) __builtin_shufflevector(lo, hi,                         \
        0,1,2,3,4,5,6,7,8,9,10,11,12,13,14,15)

// load 4 A fragments (4 row-tiles) for K-chunk starting at element kkv
#define LOAD_A4(dst, kkv)                                                     \
    do {                                                                      \
        const __bf16* ap_ = lx + (kkv);                                       \
        dst[0] = CAT16(*(const bf16x8*)(ap_),                                 \
                       *(const bf16x8*)(ap_ + 16));                           \
        dst[1] = CAT16(*(const bf16x8*)(ap_ + 16 * DPAD),                     \
                       *(const bf16x8*)(ap_ + 16 * DPAD + 16));               \
        dst[2] = CAT16(*(const bf16x8*)(ap_ + 32 * DPAD),                     \
                       *(const bf16x8*)(ap_ + 32 * DPAD + 16));               \
        dst[3] = CAT16(*(const bf16x8*)(ap_ + 48 * DPAD),                     \
                       *(const bf16x8*)(ap_ + 48 * DPAD + 16));               \
    } while (0)

#define WMMA4(Acc, Afrag, Bf)                                                 \
    do {                                                                      \
        Acc[0] = __builtin_amdgcn_wmma_f32_16x16x32_bf16(                     \
            false, Afrag[0], false, Bf, (short)0, Acc[0], false, false);      \
        Acc[1] = __builtin_amdgcn_wmma_f32_16x16x32_bf16(                     \
            false, Afrag[1], false, Bf, (short)0, Acc[1], false, false);      \
        Acc[2] = __builtin_amdgcn_wmma_f32_16x16x32_bf16(                     \
            false, Afrag[2], false, Bf, (short)0, Acc[2], false, false);      \
        Acc[3] = __builtin_amdgcn_wmma_f32_16x16x32_bf16(                     \
            false, Afrag[3], false, Bf, (short)0, Acc[3], false, false);      \
    } while (0)

// =====================================================================
// Kernel A: cb = embedding_sum / max(usage,1e-8); emit f32 + bf16 + cb_sq
// =====================================================================
__global__ __launch_bounds__(128)
void vq_codebook_kernel(const float* __restrict__ es,
                        const float* __restrict__ usage,
                        float* __restrict__ cb_f32,
                        __bf16* __restrict__ cb_bf16,
                        float* __restrict__ cb_sq,
                        float* __restrict__ loss_ptr) {
    const int k = blockIdx.x;
    const int t = threadIdx.x;
    const float inv = 1.0f / fmaxf(usage[k], 1e-8f);
    const long long base = (long long)k * DIM;

    float s = 0.0f;
#pragma unroll
    for (int i = 0; i < 4; ++i) {
        const int d = t + i * 128;
        float v = es[base + d] * inv;
        cb_f32[base + d]  = v;
        cb_bf16[base + d] = (__bf16)v;
        s = fmaf(v, v, s);
    }
    __shared__ float red[4];
#pragma unroll
    for (int off = 16; off >= 1; off >>= 1) s += __shfl_down(s, off, 32);
    if ((t & 31) == 0) red[t >> 5] = s;
    __syncthreads();
    if (t == 0) {
        cb_sq[k] = red[0] + red[1] + red[2] + red[3];
        if (k == 0) *loss_ptr = 0.0f;   // re-zero loss every launch
    }
}

// =====================================================================
// Kernel B: fused bf16-WMMA distance GEMM + per-row argmin
//   score(n,k) = cb_sq[k] - 2 * dot(x[n], cb[k])
//   block = 256 threads = 8 waves; block owns 64 x rows (4 tiles/wave);
//   each wave sweeps 16-column codebook tiles: K/(8*16) = 32 iterations.
//   B operand: double-buffered global_load_b128 straight from L2.
//   A operand: double-buffered ds_load_b128 (issued one half-chunk ahead).
// =====================================================================
__global__ __launch_bounds__(256)
void vq_argmin_kernel(const float* __restrict__ x,
                      const __bf16* __restrict__ cbh,
                      const float* __restrict__ cbsq,
                      int*   __restrict__ out_idx,
                      float* __restrict__ out_idx_f) {
    extern __shared__ __align__(16) char smem[];
    __bf16* lds_x    = (__bf16*)smem;
    float*  lds_cbsq = (float*)(smem + SM_X);
    float*  red_best = (float*)(smem + SM_X + SM_CBSQ);
    int*    red_idx  = (int*)  (smem + SM_X + SM_CBSQ + SM_RBEST);

    const int tid   = threadIdx.x;
    const int lane  = tid & 31;
    const int wave  = tid >> 5;
    const int l15   = lane & 15;
    const int lhalf = lane >> 4;              // 0 or 1
    const long long rowBase = (long long)blockIdx.x * BM;

    // ---- stage 64x512 x tile into LDS as bf16 (16B packed stores) ----
    {
        const int r  = tid >> 2;              // 0..63
        const int c0 = (tid & 3) * 128;       // 0..384
        const float*  xp = x + (rowBase + r) * DIM + c0;
        __bf16*       lp = lds_x + r * DPAD + c0;
#pragma unroll
        for (int i = 0; i < 128; i += 8) {
            float4 va = *(const float4*)(xp + i);
            float4 vb = *(const float4*)(xp + i + 4);
            bf16x8 h = { (__bf16)va.x, (__bf16)va.y, (__bf16)va.z, (__bf16)va.w,
                         (__bf16)vb.x, (__bf16)vb.y, (__bf16)vb.z, (__bf16)vb.w };
            *(bf16x8*)(lp + i) = h;
        }
    }
    for (int i = tid; i < KCODES; i += 256) lds_cbsq[i] = cbsq[i];
    __syncthreads();

    float best[4][8]; int bidx[4][8];
#pragma unroll
    for (int t2 = 0; t2 < 4; ++t2)
#pragma unroll
        for (int i = 0; i < 8; ++i) { best[t2][i] = 3.4e38f; bidx[t2][i] = 0; }

    const __bf16* lx = lds_x + l15 * DPAD + lhalf * 8;

    for (int it = 0; it < KCODES / (8 * 16); ++it) {
        const int colTile = (it * 8 + wave) * 16;
        const float cbsq_l = lds_cbsq[colTile + l15];
        v8f acc[4];
        acc[0] = (v8f){0.f,0.f,0.f,0.f,0.f,0.f,0.f,0.f};
        acc[1] = acc[0]; acc[2] = acc[0]; acc[3] = acc[0];

        // per-lane base of this wave's 16 codebook rows (+ lane-half offset)
        const __bf16* bline = cbh + ((long long)colTile + l15) * DIM + lhalf * 8;

        bf16x8 p0a, p0b, p1a, p1b;
        v16bf Aa[4], Ab[4];
        TR16_ISSUE2(p0a, p0b, bline, 0);                  // B chunk kk=0
        LOAD_A4(Aa, 0);                                   // A chunk kk=0

#pragma unroll
        for (int kk = 0; kk < DIM; kk += 64) {
            // prefetch kk+32 (B into buf1, A into Ab), then consume kk
            TR16_ISSUE2(p1a, p1b, bline, (kk + 32) * 2);
            LOAD_A4(Ab, kk + 32);
            TR16_WAIT2(p0a, p0b, "0x2");
            {
                v16bf Bf = CAT16(p0a, p0b);
                WMMA4(acc, Aa, Bf);
            }
            // prefetch kk+64 (if any), then consume kk+32
            if (kk + 64 < DIM) {
                TR16_ISSUE2(p0a, p0b, bline, (kk + 64) * 2);
                LOAD_A4(Aa, kk + 64);
                TR16_WAIT2(p1a, p1b, "0x2");
            } else {
                TR16_WAIT2(p1a, p1b, "0x0");
            }
            {
                v16bf Bf = CAT16(p1a, p1b);
                WMMA4(acc, Ab, Bf);
            }
        }

        // ---- fold this tile's scores into running argmin ----
        const int col = colTile + l15;
#pragma unroll
        for (int t2 = 0; t2 < 4; ++t2)
#pragma unroll
            for (int i = 0; i < 8; ++i) {
                float s = fmaf(-2.0f, acc[t2][i], cbsq_l);
                if (s < best[t2][i]) { best[t2][i] = s; bidx[t2][i] = col; }
            }
    }

    // ---- cross-lane argmin (each row lives across a 16-lane half) ----
#pragma unroll
    for (int t2 = 0; t2 < 4; ++t2)
#pragma unroll
        for (int i = 0; i < 8; ++i) {
            float s = best[t2][i]; int ix = bidx[t2][i];
#pragma unroll
            for (int off = 8; off >= 1; off >>= 1) {
                float os = __shfl_xor(s, off, 16);
                int   oi = __shfl_xor(ix, off, 16);
                if (os < s || (os == s && oi < ix)) { s = os; ix = oi; }
            }
            best[t2][i] = s; bidx[t2][i] = ix;
        }

    if (l15 == 0) {
#pragma unroll
        for (int t2 = 0; t2 < 4; ++t2)
#pragma unroll
            for (int i = 0; i < 8; ++i) {
                red_best[wave * BM + t2 * 16 + lhalf * 8 + i] = best[t2][i];
                red_idx [wave * BM + t2 * 16 + lhalf * 8 + i] = bidx[t2][i];
            }
    }
    __syncthreads();

    // ---- cross-wave argmin + write indices ----
    if (tid < BM) {
        float bs = red_best[tid]; int bi = red_idx[tid];
#pragma unroll
        for (int w = 1; w < 8; ++w) {
            float os = red_best[w * BM + tid]; int oi = red_idx[w * BM + tid];
            if (os < bs || (os == bs && oi < bi)) { bs = os; bi = oi; }
        }
        out_idx  [rowBase + tid] = bi;
        out_idx_f[rowBase + tid] = (float)bi;
    }
}

// =====================================================================
// Kernel C: quantized = cb[idx]; commitment loss = mean((x - q)^2)
// =====================================================================
__global__ __launch_bounds__(128)
void vq_gather_kernel(const float* __restrict__ x,
                      const float* __restrict__ cb,
                      const int*   __restrict__ idxs,
                      float* __restrict__ out,
                      float* __restrict__ loss_ptr) {
    const int n = blockIdx.x;
    const int t = threadIdx.x;
    const int idx = idxs[n];

    const float4* qv = (const float4*)(cb + (long long)idx * DIM);
    const float4* xv = (const float4*)(x  + (long long)n   * DIM);
    float4*       ov = (float4*)(out + (long long)n * DIM);

    float4 q = qv[t];
    float4 xx = xv[t];
    ov[t] = q;

    float dx = xx.x - q.x, dy = xx.y - q.y, dz = xx.z - q.z, dw = xx.w - q.w;
    float s = dx*dx + dy*dy + dz*dz + dw*dw;

    __shared__ float red[4];
#pragma unroll
    for (int off = 16; off >= 1; off >>= 1) s += __shfl_down(s, off, 32);
    if ((t & 31) == 0) red[t >> 5] = s;
    __syncthreads();
    if (t == 0) {
        float tot = red[0] + red[1] + red[2] + red[3];
        atomicAdd(loss_ptr, tot * (1.0f / ((float)NROWS * (float)DIM)));
    }
}

// =====================================================================
// launch
// =====================================================================
extern "C" void kernel_launch(void* const* d_in, const int* in_sizes, int n_in,
                              void* d_out, int out_size, void* d_ws, size_t ws_size,
                              hipStream_t stream) {
    const float* x     = (const float*)d_in[0];   // [8,4096,512] f32
    const float* es    = (const float*)d_in[1];   // [4096,512]   f32
    const float* usage = (const float*)d_in[2];   // [4096]       f32

    float* out      = (float*)d_out;                       // [8,4096,512]
    float* outIdxF  = out + (size_t)NROWS * DIM;           // [8,4096] (as float)
    float* lossPtr  = outIdxF + NROWS;                     // scalar

    char* ws = (char*)d_ws;
    float*  ws_cb   = (float*)(ws);                                    // 8 MiB
    __bf16* ws_cbh  = (__bf16*)(ws + (size_t)8 * 1024 * 1024);         // 4 MiB
    float*  ws_cbsq = (float*)(ws + (size_t)12 * 1024 * 1024);         // 16 KiB
    int*    ws_idx  = (int*)  (ws + (size_t)12 * 1024 * 1024 + 65536); // 128 KiB

    vq_codebook_kernel<<<dim3(KCODES), dim3(128), 0, stream>>>(
        es, usage, ws_cb, ws_cbh, ws_cbsq, lossPtr);

    vq_argmin_kernel<<<dim3(NROWS / BM), dim3(256), SM_TOTAL, stream>>>(
        x, ws_cbh, ws_cbsq, ws_idx, outIdxF);

    vq_gather_kernel<<<dim3(NROWS), dim3(128), 0, stream>>>(
        x, ws_cb, ws_idx, out, lossPtr);
}
